// DecLayer_54357106098674
// MI455X (gfx1250) — matrix operand order repaired and checked by reading, not compiled, not run on hardware
//
#include <hip/hip_runtime.h>
#include <hip/hip_bf16.h>

// ---------------------------------------------------------------------------
// MI455X / gfx1250 implementation of the neighbor-MLP decoder layer.
// bf16 WMMA (v_wmma_f32_16x16x32_bf16) everywhere, f32 accumulate.
// 4 nodes per workgroup (M=128), 8 waves x 16-row M-tiles, weights streamed
// as B-fragments from L2/WGP$, h_E read once from HBM and converted in-regs.
// gelu via exp2/rcp (2 hardware transcendentals) to stay under the VALU roof.
// ---------------------------------------------------------------------------

typedef __attribute__((ext_vector_type(16))) __bf16 v16bf;
typedef __attribute__((ext_vector_type(8)))  __bf16 v8bf;
typedef __attribute__((ext_vector_type(8)))  float  v8f;

#define DL_B   8
#define DL_N   4096
#define DL_K   32
#define DL_H   128
#define DL_NI  384
#define DL_FF  512          // 4*H
#define DL_SCALE 30.0f
#define DL_EPS   1e-5f

#define MST 136             // LDS row stride for 128-wide bf16 tiles (pad -> no bank conflicts)
#define FST 520             // LDS row stride for 512-wide bf16 tiles

__device__ __forceinline__ v8f wmma_bf16(v16bf a, v16bf b, v8f c) {
  // 8 args: (neg_a, A, neg_b, B, c_mod, C, reuse_a, reuse_b)
  return __builtin_amdgcn_wmma_f32_16x16x32_bf16(false, a, false, b, (short)0, c, false, false);
}

__device__ __forceinline__ v8f zero8() {
  v8f z;
  #pragma unroll
  for (int i = 0; i < 8; ++i) z[i] = 0.0f;
  return z;
}

// tanh-form gelu: x * sigmoid(2*y), y = sqrt(2/pi)*(x + 0.044715 x^3)
// = x * rcp(1 + exp2(-2*log2(e)*sqrt(2/pi)*(x + 0.044715 x^3)))
// 8 VALU ops, 2 transcendentals (v_exp_f32, v_rcp_f32).
__device__ __forceinline__ float gelu_f(float x) {
  float u = x * x;
  float p = __builtin_fmaf(u, 0.044715f, 1.0f);
  float t = x * p * -2.3021858540f;          // -2*log2(e)*sqrt(2/pi)
  float e = __builtin_amdgcn_exp2f(t);
  return x * __builtin_amdgcn_rcpf(1.0f + e);
}

__device__ __forceinline__ v16bf combine16(v8bf lo, v8bf hi) {
  union { v16bf v; v8bf h[2]; } u;
  u.h[0] = lo; u.h[1] = hi;
  return u.v;
}

// B fragment (32x16, table k order: 8h+s, 16+8h+s per lane) from row-major W[n, k] (ld elems)
__device__ __forceinline__ v16bf load_bfrag_g(const __bf16* W, int ld, int n, int k0) {
  const __bf16* p = W + (size_t)n * ld + (k0 & ~31) + ((k0 >> 4) & 1) * 8;
  return combine16(*(const v8bf*)p, *(const v8bf*)(p + 16));
}

// A fragment (16x32) from a bf16 LDS tile (row-major, ld elems)
__device__ __forceinline__ v16bf load_afrag_lds(const __bf16* buf, int row, int ld, int kbase) {
  const __bf16* p = buf + row * ld + kbase;
  return combine16(*(const v8bf*)p, *(const v8bf*)(p + 16));
}

// A fragment (16x32) directly from row-major fp32 global memory, cvt->bf16
__device__ __forceinline__ v16bf load_afrag_f32(const float* rowp, int kbase) {
  const float4 f0 = *(const float4*)(rowp + kbase);
  const float4 f1 = *(const float4*)(rowp + kbase + 4);
  const float4 f2 = *(const float4*)(rowp + kbase + 16);
  const float4 f3 = *(const float4*)(rowp + kbase + 20);
  v16bf a;
  a[0]=(__bf16)f0.x;  a[1]=(__bf16)f0.y;  a[2]=(__bf16)f0.z;  a[3]=(__bf16)f0.w;
  a[4]=(__bf16)f1.x;  a[5]=(__bf16)f1.y;  a[6]=(__bf16)f1.z;  a[7]=(__bf16)f1.w;
  a[8]=(__bf16)f2.x;  a[9]=(__bf16)f2.y;  a[10]=(__bf16)f2.z; a[11]=(__bf16)f2.w;
  a[12]=(__bf16)f3.x; a[13]=(__bf16)f3.y; a[14]=(__bf16)f3.z; a[15]=(__bf16)f3.w;
  return a;
}

// ------------------------------ prep kernels -------------------------------

__global__ void k_cvt_bf16(const float* __restrict__ s, __bf16* __restrict__ d, int n) {
  int i = blockIdx.x * 256 + threadIdx.x;
  if (i < n) d[i] = (__bf16)s[i];
}

__global__ void k_split_w1(const float* __restrict__ s, __bf16* __restrict__ w1v,
                           __bf16* __restrict__ w1e) {
  int i = blockIdx.x * 256 + threadIdx.x;
  if (i >= DL_H * (DL_H + DL_NI)) return;
  int o = i >> 9;          // / 512
  int c = i & 511;
  float v = s[i];
  if (c < DL_H) w1v[o * DL_H + c] = (__bf16)v;
  else          w1e[o * DL_NI + (c - DL_H)] = (__bf16)v;
}

// ------------------------------- main kernel -------------------------------
// grid = (B*N)/4 blocks of 256 threads (8 waves). Block g owns nodes 4g..4g+3.

__launch_bounds__(256)
__global__ void dec_layer_main(
    const float* __restrict__ hV, const float* __restrict__ hE,
    const float* __restrict__ maskV, const float* __restrict__ maskA,
    const float* __restrict__ W1b, const float* __restrict__ W2b,
    const float* __restrict__ W3b,
    const float* __restrict__ ln1g, const float* __restrict__ ln1b,
    const float* __restrict__ ln2g, const float* __restrict__ ln2b,
    const float* __restrict__ Winb, const float* __restrict__ Woutb,
    const __bf16* __restrict__ W1v, const __bf16* __restrict__ W1e,
    const __bf16* __restrict__ W2,  const __bf16* __restrict__ W3,
    const __bf16* __restrict__ Win, const __bf16* __restrict__ Wout,
    float* __restrict__ out)
{
  const int g    = blockIdx.x;
  const int tid  = threadIdx.x;
  const int wave = tid >> 5;
  const int lane = tid & 31;
  const int half = lane >> 4;       // which 16-lane half
  const int l16  = lane & 15;

  __shared__ __attribute__((aligned(16))) __bf16 mbuf[128 * MST];  // 34816 B
  __shared__ __attribute__((aligned(16))) __bf16 rowbuf[16 * MST]; //  4352 B
  __shared__ __attribute__((aligned(16))) __bf16 ffbuf[16 * FST];  // 16640 B
  __shared__ __attribute__((aligned(16))) float  hvp[4 * 128];     // hv_part, reused as hpre2
  __shared__ __attribute__((aligned(16))) float  dhb[4 * 128];
  __shared__ __attribute__((aligned(16))) float  hVf[4 * 128];
  __shared__ __attribute__((aligned(16))) float  h1f[4 * 128];

  // ---- init: zero pads, stage h_V (f32 residual + bf16 A-tile rows 0..3) ----
  for (int i = tid; i < 16 * MST; i += 256) rowbuf[i] = (__bf16)0.0f;
  for (int i = tid; i < 16 * FST; i += 256) ffbuf[i]  = (__bf16)0.0f;
  for (int i = tid; i < 512; i += 256) dhb[i] = 0.0f;
  for (int e = tid; e < 512; e += 256) {
    int nd = e >> 7, c = e & 127;
    float v = hV[(size_t)(g * 4 + nd) * DL_H + c];
    hVf[e] = v;
    rowbuf[nd * MST + c] = (__bf16)v;
  }
  __syncthreads();

  // ---- Stage A: hv_part[node, :] = h_V @ W1v^T   (broadcast-row WMMA) ----
  {
    v8f acc = zero8();
    #pragma unroll
    for (int kk = 0; kk < 4; ++kk) {
      v16bf a = load_afrag_lds(rowbuf, l16, MST, kk * 32 + half * 8);
      v16bf b = load_bfrag_g(W1v, DL_H, wave * 16 + l16, kk * 32 + half * 16);
      acc = wmma_bf16(a, b, acc);
    }
    if (half == 0) {
      int col = wave * 16 + l16;
      #pragma unroll
      for (int v = 0; v < 4; ++v) hvp[v * 128 + col] = acc[v];
    }
  }
  __syncthreads();

  const int myrow = wave * 16 + l16;                 // 0..127 within node group
  const int nd    = wave >> 1;                       // node (0..3) owned by this wave
  const float* Erow = hE + ((size_t)g * 128 + myrow) * DL_NI;

  // ---- Stage B: m1 = gelu(hE @ W1e^T + hv_part + b1)   M=128,N=128,K=384 ----
  {
    v8f acc[8];
    // peeled kk = 0: C is the inline-0 operand
    {
      v16bf a = load_afrag_f32(Erow, half * 8);
      #pragma unroll
      for (int nt = 0; nt < 8; ++nt) {
        v16bf b = load_bfrag_g(W1e, DL_NI, nt * 16 + l16, half * 16);
        acc[nt] = wmma_bf16(a, b, zero8());
      }
    }
    for (int kk = 1; kk < 12; ++kk) {
      v16bf a = load_afrag_f32(Erow, kk * 32 + half * 8);
      #pragma unroll
      for (int nt = 0; nt < 8; ++nt) {
        v16bf b = load_bfrag_g(W1e, DL_NI, nt * 16 + l16, kk * 32 + half * 16);
        acc[nt] = wmma_bf16(a, b, acc[nt]);
      }
    }
    const int rbase = wave * 16 + half * 8;
    #pragma unroll
    for (int nt = 0; nt < 8; ++nt) {
      int col = nt * 16 + l16;
      float bias = hvp[nd * 128 + col] + W1b[col];
      #pragma unroll
      for (int v = 0; v < 8; ++v)
        mbuf[(rbase + v) * MST + col] = (__bf16)gelu_f(acc[nt][v] + bias);
    }
  }

  // ---- Stage C: m2 = gelu(m1 @ W2^T + b2)   M=128,N=128,K=128 ----
  {
    v8f acc[8];
    #pragma unroll
    for (int nt = 0; nt < 8; ++nt) acc[nt] = zero8();

    #pragma unroll
    for (int kk = 0; kk < 4; ++kk) {
      v16bf a = load_afrag_lds(mbuf, myrow, MST, kk * 32 + half * 8);
      #pragma unroll
      for (int nt = 0; nt < 8; ++nt) {
        v16bf b = load_bfrag_g(W2, DL_H, nt * 16 + l16, kk * 32 + half * 16);
        acc[nt] = wmma_bf16(a, b, acc[nt]);
      }
    }
    const int rbase = wave * 16 + half * 8;
    #pragma unroll
    for (int nt = 0; nt < 8; ++nt) {
      int col = nt * 16 + l16;
      float bias = W2b[col];
      #pragma unroll
      for (int v = 0; v < 8; ++v)
        mbuf[(rbase + v) * MST + col] = (__bf16)gelu_f(acc[nt][v] + bias);
    }
  }

  // ---- Stage D: m3 = m2 @ W3^T + b3 ; masked K-sum -> dh ----
  {
    v8f acc[8];
    #pragma unroll
    for (int nt = 0; nt < 8; ++nt) acc[nt] = zero8();

    #pragma unroll
    for (int kk = 0; kk < 4; ++kk) {
      v16bf a = load_afrag_lds(mbuf, myrow, MST, kk * 32 + half * 8);
      #pragma unroll
      for (int nt = 0; nt < 8; ++nt) {
        v16bf b = load_bfrag_g(W3, DL_H, nt * 16 + l16, kk * 32 + half * 16);
        acc[nt] = wmma_bf16(a, b, acc[nt]);
      }
    }
    // attention mask for the 8 k-rows held by this lane
    float mk[8];
    {
      const int lb = (wave & 1) * 16 + half * 8;     // local k within node
      const float* mp = maskA + (size_t)g * 128 + nd * 32 + lb;
      float4 m0 = *(const float4*)mp;
      float4 m1 = *(const float4*)(mp + 4);
      mk[0]=m0.x; mk[1]=m0.y; mk[2]=m0.z; mk[3]=m0.w;
      mk[4]=m1.x; mk[5]=m1.y; mk[6]=m1.z; mk[7]=m1.w;
    }
    #pragma unroll
    for (int nt = 0; nt < 8; ++nt) {
      int col = nt * 16 + l16;
      float b3 = W3b[col];
      float s = 0.0f;
      #pragma unroll
      for (int v = 0; v < 8; ++v) s += (acc[nt][v] + b3) * mk[v];
      s += __shfl_xor(s, 16, 32);                    // combine the two lane halves
      if (half == 0) atomicAdd(&dhb[nd * 128 + col], s);   // ds_add_f32
    }
  }
  __syncthreads();

  // ---- LN1: h = LN(h_V + dh/SCALE); stash f32 residual + bf16 A-rows ----
  if (wave < 4) {
    const int ndl = wave;
    float x[4], s = 0.0f;
    #pragma unroll
    for (int j = 0; j < 4; ++j) {
      int c = lane + j * 32;
      x[j] = hVf[ndl * 128 + c] + dhb[ndl * 128 + c] * (1.0f / DL_SCALE);
      s += x[j];
    }
    #pragma unroll
    for (int o = 16; o >= 1; o >>= 1) s += __shfl_xor(s, o, 32);
    const float mean = s * (1.0f / 128.0f);
    float vv = 0.0f;
    #pragma unroll
    for (int j = 0; j < 4; ++j) { float d = x[j] - mean; vv += d * d; }
    #pragma unroll
    for (int o = 16; o >= 1; o >>= 1) vv += __shfl_xor(vv, o, 32);
    const float rs = rsqrtf(vv * (1.0f / 128.0f) + DL_EPS);
    #pragma unroll
    for (int j = 0; j < 4; ++j) {
      int c = lane + j * 32;
      float y = (x[j] - mean) * rs * ln1g[c] + ln1b[c];
      h1f[ndl * 128 + c] = y;
      rowbuf[ndl * MST + c] = (__bf16)y;
    }
  }
  __syncthreads();

  // ---- Stage E: ff1 = gelu(h @ Win^T + bin)  (broadcast-row, N=512) ----
  {
    v8f acc[4];
    #pragma unroll
    for (int t = 0; t < 4; ++t) acc[t] = zero8();

    #pragma unroll
    for (int kk = 0; kk < 4; ++kk) {
      v16bf a = load_afrag_lds(rowbuf, l16, MST, kk * 32 + half * 8);
      #pragma unroll
      for (int t = 0; t < 4; ++t) {
        int n = (wave * 4 + t) * 16 + l16;
        v16bf b = load_bfrag_g(Win, DL_H, n, kk * 32 + half * 16);
        acc[t] = wmma_bf16(a, b, acc[t]);
      }
    }
    if (half == 0) {
      #pragma unroll
      for (int t = 0; t < 4; ++t) {
        int col = (wave * 4 + t) * 16 + l16;
        float bi = Winb[col];
        #pragma unroll
        for (int v = 0; v < 4; ++v)
          ffbuf[v * FST + col] = (__bf16)gelu_f(acc[t][v] + bi);
      }
    }
  }
  __syncthreads();

  // ---- Stage F: ff2 = ff1 @ Wout^T + bout ; hpre2 = h + ff2 ----
  {
    v8f acc;
    // peeled kk = 0
    {
      v16bf a = load_afrag_lds(ffbuf, l16, FST, half * 8);
      v16bf b = load_bfrag_g(Wout, DL_FF, wave * 16 + l16, half * 16);
      acc = wmma_bf16(a, b, zero8());
    }
    for (int kk = 1; kk < 16; ++kk) {
      v16bf a = load_afrag_lds(ffbuf, l16, FST, kk * 32 + half * 8);
      v16bf b = load_bfrag_g(Wout, DL_FF, wave * 16 + l16, kk * 32 + half * 16);
      acc = wmma_bf16(a, b, acc);
    }
    if (half == 0) {
      int col = wave * 16 + l16;
      float bo = Woutb[col];
      #pragma unroll
      for (int v = 0; v < 4; ++v)
        hvp[v * 128 + col] = h1f[v * 128 + col] + acc[v] + bo;  // reuse hvp as hpre2
    }
  }
  __syncthreads();

  // ---- LN2 + output mask ----
  if (wave < 4) {
    const int ndl = wave;
    float x[4], s = 0.0f;
    #pragma unroll
    for (int j = 0; j < 4; ++j) {
      int c = lane + j * 32;
      x[j] = hvp[ndl * 128 + c];
      s += x[j];
    }
    #pragma unroll
    for (int o = 16; o >= 1; o >>= 1) s += __shfl_xor(s, o, 32);
    const float mean = s * (1.0f / 128.0f);
    float vv = 0.0f;
    #pragma unroll
    for (int j = 0; j < 4; ++j) { float d = x[j] - mean; vv += d * d; }
    #pragma unroll
    for (int o = 16; o >= 1; o >>= 1) vv += __shfl_xor(vv, o, 32);
    const float rs = rsqrtf(vv * (1.0f / 128.0f) + DL_EPS);
    const float mv = maskV[(size_t)g * 4 + ndl];
    #pragma unroll
    for (int j = 0; j < 4; ++j) {
      int c = lane + j * 32;
      float y = (x[j] - mean) * rs * ln2g[c] + ln2b[c];
      out[(size_t)(g * 4 + ndl) * DL_H + c] = mv * y;
    }
  }
}

// ------------------------------- launcher ----------------------------------

extern "C" void kernel_launch(void* const* d_in, const int* in_sizes, int n_in,
                              void* d_out, int out_size, void* d_ws, size_t ws_size,
                              hipStream_t stream) {
  const float* hV    = (const float*)d_in[0];
  const float* hE    = (const float*)d_in[1];
  const float* maskV = (const float*)d_in[2];
  const float* maskA = (const float*)d_in[3];
  const float* W1w   = (const float*)d_in[4];
  const float* W1b   = (const float*)d_in[5];
  const float* W2w   = (const float*)d_in[6];
  const float* W2b   = (const float*)d_in[7];
  const float* W3w   = (const float*)d_in[8];
  const float* W3b   = (const float*)d_in[9];
  const float* ln1g  = (const float*)d_in[10];
  const float* ln1b  = (const float*)d_in[11];
  const float* ln2g  = (const float*)d_in[12];
  const float* ln2b  = (const float*)d_in[13];
  const float* Winw  = (const float*)d_in[14];
  const float* Winb  = (const float*)d_in[15];
  const float* Woutw = (const float*)d_in[16];
  const float* Woutb = (const float*)d_in[17];

  __bf16* ws   = (__bf16*)d_ws;
  __bf16* W1v  = ws;                 // 128*128
  __bf16* W1e  = W1v + 16384;        // 128*384
  __bf16* W2   = W1e + 49152;        // 128*128
  __bf16* W3   = W2  + 16384;        // 128*128
  __bf16* Win  = W3  + 16384;        // 512*128
  __bf16* Wout = Win + 65536;        // 128*512

  // weight conversion (fp32 -> bf16), deterministic each call
  k_split_w1<<<256, 256, 0, stream>>>(W1w, W1v, W1e);
  k_cvt_bf16<<<64, 256, 0, stream>>>(W2w, W2, 16384);
  k_cvt_bf16<<<64, 256, 0, stream>>>(W3w, W3, 16384);
  k_cvt_bf16<<<256, 256, 0, stream>>>(Winw, Win, 65536);
  k_cvt_bf16<<<256, 256, 0, stream>>>(Woutw, Wout, 65536);

  const int ngroups = (DL_B * DL_N) / 4;   // 8192 workgroups, 4 nodes each
  dec_layer_main<<<ngroups, 256, 0, stream>>>(
      hV, hE, maskV, maskA,
      W1b, W2b, W3b, ln1g, ln1b, ln2g, ln2b, Winb, Woutb,
      W1v, W1e, W2, W3, Win, Wout,
      (float*)d_out);
}
